// SGLC_Classifier_37546604101923
// MI455X (gfx1250) — matrix operand order repaired and checked by reading, not compiled, not run on hardware
//
#include <hip/hip_runtime.h>
#include <hip/hip_bf16.h>
#include <cstdint>

typedef __attribute__((ext_vector_type(16))) _Float16 v16h;
typedef __attribute__((ext_vector_type(8)))  float    v8f;

#define SEQ    128
#define BATCH  128
#define NNODE  19
#define DIM    100
#define NHEAD  16
#define GSC    0.3f
#define STEPS  5
#define EPS    1e-8f

static __device__ __forceinline__ v8f wmma_f16(v16h a, v16h b, v8f c) {
  // D = A(16x32 f16) * B(32x16 f16) + C(16x16 f32)
  return __builtin_amdgcn_wmma_f32_16x16x32_f16(false, a, false, b, (short)0, c, false, false);
}

// A-matrix fragment (16x32 f16), row-major source [rows][ld].
// Lane holds row (row0 + lane%16). Lanes 0-15 hold K {0..15},
// lanes 16-31 hold K {16..31}; two contiguous 8-half (16B) reads.
static __device__ __forceinline__ v16h load_fragA(const _Float16* base, int row0, int ld,
                                                  int kbase, int lane) {
  const _Float16* r = base + (size_t)(row0 + (lane & 15)) * ld + kbase + ((lane >> 4) << 4);
  v16h a;
#pragma unroll
  for (int v = 0; v < 8; ++v) a[v] = r[v];
#pragma unroll
  for (int v = 0; v < 8; ++v) a[8 + v] = r[8 + v];
  return a;
}

// B fragment from a pre-swizzled matrix: element ((c*7+nt)*32 + lane)*16 + v
// holds B[k = c*32 + 16*(lane>>4) + v][n = nt*16 + lane%16].
// 16 contiguous halfs per lane -> two b128 loads.
static __device__ __forceinline__ v16h load_fragB_swz(const _Float16* Bsw, int c, int nt,
                                                      int lane) {
  const _Float16* p = Bsw + ((size_t)(c * 7 + nt) * 32 + lane) * 16;
  v16h b;
#pragma unroll
  for (int v = 0; v < 16; ++v) b[v] = p[v];
  return b;
}

// B fragment for B = Y^T with Y row-major [rows][ld]: B(k,n) = Y[n][k].
static __device__ __forceinline__ v16h load_fragBT(const _Float16* Y, int kbase, int col0,
                                                   int ld, int lane) {
  const _Float16* r = Y + (size_t)(col0 + (lane & 15)) * ld + kbase + ((lane >> 4) << 4);
  v16h b;
#pragma unroll
  for (int v = 0; v < 16; ++v) b[v] = r[v];
  return b;
}

// Accumulate A[32 x 32*kchunks] @ Bswz tile (mt,nt) into acc.
static __device__ __forceinline__ v8f mm_acc(const _Float16* A, int ldA, int kchunks,
                                             const _Float16* Bsw,
                                             int mt, int nt, int lane, v8f acc) {
  for (int c = 0; c < kchunks; ++c) {
    v16h a = load_fragA(A, mt * 16, ldA, c * 32, lane);
    v16h b = load_fragB_swz(Bsw, c, nt, lane);
    acc = wmma_f16(a, b, acc);
  }
  return acc;
}

// Async global->LDS copy of one 7600B x-tile (475 x 16B chunks), ASYNCcnt-tracked.
static __device__ __forceinline__ void issue_async_x(const float* xg, float* xdst, int tid,
                                                     int nthreads) {
  const char* g = (const char*)xg;
  unsigned lbase = (unsigned)(uintptr_t)xdst;  // generic LDS addr truncates to LDS offset
  for (int c = tid; c < 475; c += nthreads) {
    asm volatile("global_load_async_to_lds_b128 %0, %1, off"
                 :: "v"(lbase + (unsigned)(c * 16)), "v"(g + (size_t)c * 16)
                 : "memory");
  }
}

static __device__ __forceinline__ void wait_async0() {
  asm volatile("s_wait_asynccnt 0x0" ::: "memory");
}

// ---------------------------------------------------------------------------
// Kernel 1: convert the 7 GGNN weight matrices (100x100 f32) to f16 in
// fragment-swizzled order (4 K-chunks x 7 N-tiles x 32 lanes x 16 halfs) so
// B-fragment loads are contiguous b128s.
// ---------------------------------------------------------------------------
__global__ void k_convert_weights(const float* W0, const float* W1, const float* W2,
                                  const float* W3, const float* W4, const float* W5,
                                  const float* W6, _Float16* wpad) {
  const float* src[7] = {W0, W1, W2, W3, W4, W5, W6};
  const float* s = src[blockIdx.x];
  _Float16* d = wpad + (size_t)blockIdx.x * 14336;
  for (int i = threadIdx.x; i < 14336; i += blockDim.x) {
    int v = i & 15;
    int l = (i >> 4) & 31;
    int tileidx = i >> 9;            // c*7 + nt
    int nt = tileidx % 7, c = tileidx / 7;
    int k = c * 32 + ((l >> 4) << 4) + v;
    int n = nt * 16 + (l & 15);
    d[i] = (k < DIM && n < DIM) ? (_Float16)s[k * DIM + n] : (_Float16)0.0f;
  }
}

// ---------------------------------------------------------------------------
// Kernel 2: sequential graph-learner recurrence, one block per batch element.
// x-tiles are double-buffered via global_load_async_to_lds_b128 so the t+1
// HBM read overlaps the t Gram/WMMA compute. Gram Y*Y^T over 50 K-chunks is
// split over 4 waves with an LDS partial reduce; then relu + row-normalize,
//    adj = GSC*adj + (1-GSC)*att.
// ---------------------------------------------------------------------------
__global__ void k_graph_adj(const float* __restrict__ inputs,
                            const float* __restrict__ supports,
                            const float* __restrict__ glw,
                            float* __restrict__ adjG) {
  extern __shared__ char smem[];
  _Float16* Y   = (_Float16*)smem;                 // [32][1600] f16   102400 B
  float* xs0    = (float*)(smem + 102400);         // [19][100] f32      7616 B
  float* xs1    = (float*)(smem + 110016);         // [19][100] f32      7616 B
  float* wl     = (float*)(smem + 117632);         // [16][100] f32      6400 B
  float* attP   = (float*)(smem + 124032);         // [4][32][32] f32   16384 B
  float* att    = (float*)(smem + 140416);         // [19*19] f32        1536 B
  float* rowsum = (float*)(smem + 141952);         // [19] f32            128 B
  float* adjL   = (float*)(smem + 142080);         // [19*19] f32        1536 B

  const int tid  = threadIdx.x;
  const int b    = blockIdx.x;
  const int wave = tid >> 5, lane = tid & 31;
  float* xsbuf[2] = {xs0, xs1};

  for (int i = tid; i < NHEAD * DIM; i += blockDim.x) wl[i] = glw[i];
  for (int i = tid; i < NNODE * NNODE; i += blockDim.x)
    adjL[i] = supports[(size_t)b * NNODE * NNODE + i];
  for (int i = tid; i < (32 - NNODE) * 1600; i += blockDim.x)  // zero pad rows 19..31 once
    Y[NNODE * 1600 + i] = (_Float16)0.0f;

  // prologue: async-stage x for t=0
  issue_async_x(inputs + (size_t)b * (NNODE * DIM), xs0, tid, blockDim.x);
  __syncthreads();

  for (int t = 0; t < SEQ; ++t) {
    wait_async0();
    __syncthreads();
    if (t + 1 < SEQ)  // prefetch next tile into the other buffer (overlaps WMMA below)
      issue_async_x(inputs + ((size_t)(t + 1) * BATCH + b) * (NNODE * DIM),
                    xsbuf[(t + 1) & 1], tid, blockDim.x);
    const float* xs = xsbuf[t & 1];

    // Y[n][h*100+d] = w[h][d]*x[n][d] / (||w[h]*x[n]|| + eps)
    for (int p = tid; p < NHEAD * NNODE; p += blockDim.x) {
      int h = p / NNODE, n = p % NNODE;
      const float* wrow = wl + h * DIM;
      const float* xrow = xs + n * DIM;
      float ss = 0.0f;
      for (int d = 0; d < DIM; ++d) { float v = wrow[d] * xrow[d]; ss += v * v; }
      float inv = 1.0f / (sqrtf(ss) + EPS);
      _Float16* yrow = Y + (size_t)n * 1600 + h * DIM;
      for (int d = 0; d < DIM; ++d) yrow[d] = (_Float16)(wrow[d] * xrow[d] * inv);
    }
    __syncthreads();

    // Gram: att32 = Y(32x1600) * Y^T over 50 K-chunks, waves 0..3 interleaved
    if (wave < 4) {
      v8f a00 = {}, a01 = {}, a10 = {}, a11 = {};
      for (int c = wave; c < 50; c += 4) {
        int kb = c * 32;
        v16h fa0 = load_fragA(Y, 0, 1600, kb, lane);
        v16h fa1 = load_fragA(Y, 16, 1600, kb, lane);
        v16h fb0 = load_fragBT(Y, kb, 0, 1600, lane);
        v16h fb1 = load_fragBT(Y, kb, 16, 1600, lane);
        a00 = wmma_f16(fa0, fb0, a00);
        a01 = wmma_f16(fa0, fb1, a01);
        a10 = wmma_f16(fa1, fb0, a10);
        a11 = wmma_f16(fa1, fb1, a11);
      }
      float* ap = attP + wave * 1024;
      int colL = lane & 15, rb = (lane >> 4) << 3;
#pragma unroll
      for (int r = 0; r < 8; ++r) {
        ap[(rb + r) * 32 + colL]           = a00[r];
        ap[(rb + r) * 32 + 16 + colL]      = a01[r];
        ap[(16 + rb + r) * 32 + colL]      = a10[r];
        ap[(16 + rb + r) * 32 + 16 + colL] = a11[r];
      }
    }
    __syncthreads();

    // reduce 4 partials, scale by 1/H, relu
    for (int p = tid; p < NNODE * NNODE; p += blockDim.x) {
      int n = p / NNODE, m = p % NNODE;
      float s = attP[n * 32 + m] + attP[1024 + n * 32 + m] +
                attP[2048 + n * 32 + m] + attP[3072 + n * 32 + m];
      s *= (1.0f / NHEAD);
      att[p] = s > 0.0f ? s : 0.0f;
    }
    __syncthreads();

    if (tid < NNODE) {
      float s = 0.0f;
      for (int m = 0; m < NNODE; ++m) s += att[tid * NNODE + m];
      rowsum[tid] = s + EPS;
    }
    __syncthreads();

    for (int p = tid; p < NNODE * NNODE; p += blockDim.x) {
      int n = p / NNODE;
      adjL[p] = GSC * adjL[p] + (1.0f - GSC) * att[p] / rowsum[n];
    }
    __syncthreads();
  }

  for (int i = tid; i < NNODE * NNODE; i += blockDim.x)
    adjG[(size_t)b * NNODE * NNODE + i] = adjL[i];
}

// ---------------------------------------------------------------------------
// Kernel 3: GGNN on the LAST time step only (only hs[-1] feeds the classifier)
// + fused classifier dot product. One block per batch element; 14 output tiles
// per GEMM over 8 waves; bias preloaded into the WMMA C operand; all B
// operands (global weights and LDS mS) are in swizzled fragment layout.
// ---------------------------------------------------------------------------
__global__ void k_ggnn_classify(const float* __restrict__ inputs,
                                const float* __restrict__ adjG,
                                const _Float16* __restrict__ wpad,
                                const float* __restrict__ bz,
                                const float* __restrict__ br,
                                const float* __restrict__ bh,
                                const float* __restrict__ Wfc,
                                const float* __restrict__ bfc,
                                float* __restrict__ out) {
  extern __shared__ char smem[];
  _Float16* adjS = (_Float16*)smem;                // [32][32]  f16 (A layout)
  _Float16* hS   = (_Float16*)(smem + 2048);       // [32][128] f16 (A layout)
  _Float16* mS   = (_Float16*)(smem + 10240);      // swizzled B: 1 chunk x 7 tiles
  _Float16* aS   = (_Float16*)(smem + 17408);      // [32][128] f16 (A layout)
  _Float16* rhS  = (_Float16*)(smem + 25600);      // [32][128] f16 (A layout)
  float*    hF   = (float*)(smem + 33792);         // [19][112] f32
  float*    zF   = (float*)(smem + 42304);         // [19][112] f32
  float*    red  = (float*)(smem + 50816);         // [256]     f32

  const int tid  = threadIdx.x, b = blockIdx.x;
  const int wave = tid >> 5, lane = tid & 31;

  const _Float16* Wmsg = wpad;
  const _Float16* Wzp  = wpad + 1 * 14336;
  const _Float16* Uzp  = wpad + 2 * 14336;
  const _Float16* Wrp  = wpad + 3 * 14336;
  const _Float16* Urp  = wpad + 4 * 14336;
  const _Float16* Whp  = wpad + 5 * 14336;
  const _Float16* Uhp  = wpad + 6 * 14336;

  if (tid < 7) __builtin_prefetch(wpad + (size_t)tid * 14336, 0, 3);

  // zero all f16 staging (adjS..rhS contiguous: 33792 B = 16896 halfs)
  {
    _Float16* z0 = (_Float16*)smem;
    for (int i = tid; i < 16896; i += blockDim.x) z0[i] = (_Float16)0.0f;
  }
  __syncthreads();

  const float* xl = inputs + ((size_t)(SEQ - 1) * BATCH + b) * (NNODE * DIM);
  for (int i = tid; i < NNODE * DIM; i += blockDim.x) {
    int n = i / DIM, d = i % DIM;
    float v = xl[i];
    hF[n * 112 + d] = v;
    hS[n * 128 + d] = (_Float16)v;
  }
  for (int p = tid; p < NNODE * NNODE; p += blockDim.x)
    adjS[(p / NNODE) * 32 + (p % NNODE)] = (_Float16)adjG[(size_t)b * NNODE * NNODE + p];
  __syncthreads();

  for (int step = 0; step < STEPS; ++step) {
    // phase 1: m = h @ Wmsg  (store m in swizzled-B layout for phase 2)
    for (int tile = wave; tile < 14; tile += 8) {
      int mt = tile / 7, nt = tile % 7;
      v8f acc = {};
      acc = mm_acc(hS, 128, 4, Wmsg, mt, nt, lane, acc);
      int col = nt * 16 + (lane & 15);
      int rb  = mt * 16 + ((lane >> 4) << 3);
      if (col < DIM) {
#pragma unroll
        for (int r = 0; r < 8; ++r) {
          int row = rb + r;
          if (row < NNODE) {
            // swizzled B index: chunk 0, nt' = col>>4, lane' = (row>>4)*16 + col%16, v = row%16
            int off = (((col >> 4) * 32) + ((row >> 4) << 4) + (col & 15)) * 16 + (row & 15);
            mS[off] = (_Float16)acc[r];
          }
        }
      }
    }
    __syncthreads();
    // phase 2: a = adj @ m   (K = 19 padded to one 32-chunk)
    for (int tile = wave; tile < 14; tile += 8) {
      int mt = tile / 7, nt = tile % 7;
      v8f acc = {};
      acc = mm_acc(adjS, 32, 1, mS, mt, nt, lane, acc);
      int col = nt * 16 + (lane & 15);
      int rb  = mt * 16 + ((lane >> 4) << 3);
      if (col < DIM) {
#pragma unroll
        for (int r = 0; r < 8; ++r) {
          int row = rb + r;
          if (row < NNODE) aS[row * 128 + col] = (_Float16)acc[r];
        }
      }
    }
    __syncthreads();
    // phase 3: z = sigmoid(a@Wz + h@Uz + bz)
    for (int tile = wave; tile < 14; tile += 8) {
      int mt = tile / 7, nt = tile % 7;
      int col = nt * 16 + (lane & 15);
      float bv = (col < DIM) ? bz[col] : 0.0f;
      v8f acc = {};
#pragma unroll
      for (int r = 0; r < 8; ++r) acc[r] = bv;
      acc = mm_acc(aS, 128, 4, Wzp, mt, nt, lane, acc);
      acc = mm_acc(hS, 128, 4, Uzp, mt, nt, lane, acc);
      int rb = mt * 16 + ((lane >> 4) << 3);
      if (col < DIM) {
#pragma unroll
        for (int r = 0; r < 8; ++r) {
          int row = rb + r;
          if (row < NNODE) zF[row * 112 + col] = 1.0f / (1.0f + expf(-acc[r]));
        }
      }
    }
    __syncthreads();
    // phase 4: r = sigmoid(a@Wr + h@Ur + br);  rh = r * h
    for (int tile = wave; tile < 14; tile += 8) {
      int mt = tile / 7, nt = tile % 7;
      int col = nt * 16 + (lane & 15);
      float bv = (col < DIM) ? br[col] : 0.0f;
      v8f acc = {};
#pragma unroll
      for (int r = 0; r < 8; ++r) acc[r] = bv;
      acc = mm_acc(aS, 128, 4, Wrp, mt, nt, lane, acc);
      acc = mm_acc(hS, 128, 4, Urp, mt, nt, lane, acc);
      int rb = mt * 16 + ((lane >> 4) << 3);
      if (col < DIM) {
#pragma unroll
        for (int r = 0; r < 8; ++r) {
          int row = rb + r;
          if (row < NNODE) {
            float rv = 1.0f / (1.0f + expf(-acc[r]));
            rhS[row * 128 + col] = (_Float16)(rv * hF[row * 112 + col]);
          }
        }
      }
    }
    __syncthreads();
    // phase 5: htil = tanh(a@Wh + rh@Uh + bh);  h = (1-z)*h + z*htil
    for (int tile = wave; tile < 14; tile += 8) {
      int mt = tile / 7, nt = tile % 7;
      int col = nt * 16 + (lane & 15);
      float bv = (col < DIM) ? bh[col] : 0.0f;
      v8f acc = {};
#pragma unroll
      for (int r = 0; r < 8; ++r) acc[r] = bv;
      acc = mm_acc(aS,  128, 4, Whp, mt, nt, lane, acc);
      acc = mm_acc(rhS, 128, 4, Uhp, mt, nt, lane, acc);
      int rb = mt * 16 + ((lane >> 4) << 3);
      if (col < DIM) {
#pragma unroll
        for (int r = 0; r < 8; ++r) {
          int row = rb + r;
          if (row < NNODE) {
            float ht = tanhf(acc[r]);
            float z  = zF[row * 112 + col];
            float hn = (1.0f - z) * hF[row * 112 + col] + z * ht;
            hF[row * 112 + col] = hn;
            hS[row * 128 + col] = (_Float16)hn;
          }
        }
      }
    }
    __syncthreads();
  }

  // classifier: out[b] = flatten(h) . Wfc + bfc
  float partial = 0.0f;
  for (int i = tid; i < NNODE * DIM; i += blockDim.x) {
    int n = i / DIM, d = i % DIM;
    partial += hF[n * 112 + d] * Wfc[i];
  }
  red[tid] = partial;
  __syncthreads();
  for (int s = 128; s > 0; s >>= 1) {
    if (tid < s) red[tid] += red[tid + s];
    __syncthreads();
  }
  if (tid == 0) out[b] = red[0] + bfc[0];
}

extern "C" void kernel_launch(void* const* d_in, const int* in_sizes, int n_in,
                              void* d_out, int out_size, void* d_ws, size_t ws_size,
                              hipStream_t stream) {
  const float* inputs   = (const float*)d_in[0];
  const float* supports = (const float*)d_in[1];
  const float* glw      = (const float*)d_in[2];
  const float* Wmsg     = (const float*)d_in[3];
  const float* Wz       = (const float*)d_in[4];
  const float* Uz       = (const float*)d_in[5];
  const float* Wr       = (const float*)d_in[6];
  const float* Ur       = (const float*)d_in[7];
  const float* Wh       = (const float*)d_in[8];
  const float* Uh       = (const float*)d_in[9];
  const float* bz       = (const float*)d_in[10];
  const float* br       = (const float*)d_in[11];
  const float* bh       = (const float*)d_in[12];
  const float* Wfc      = (const float*)d_in[13];
  const float* bfc      = (const float*)d_in[14];
  float* out = (float*)d_out;

  char* ws = (char*)d_ws;
  float*    adjG = (float*)ws;                    // B*19*19 f32 = 184832 B (256-aligned)
  _Float16* wpad = (_Float16*)(ws + 184832);      // 7*14336 f16 = 200704 B (swizzled)

  k_convert_weights<<<7, 256, 0, stream>>>(Wmsg, Wz, Uz, Wr, Ur, Wh, Uh, wpad);

  const size_t smem2 = 143616;  // Y + double-buffered x tiles + staging
  k_graph_adj<<<BATCH, 256, smem2, stream>>>(inputs, supports, glw, adjG);

  const size_t smem3 = 51840;
  k_ggnn_classify<<<BATCH, 256, smem3, stream>>>(inputs, adjG, wpad, bz, br, bh, Wfc, bfc, out);
}
